// GbootBlock_30090540875928
// MI455X (gfx1250) — matrix operand order repaired and checked by reading, not compiled, not run on hardware
//
#include <hip/hip_runtime.h>
#include <hip/hip_bf16.h>

// ---------------- problem constants ----------------
#define Bc   8
#define Nc   4096
#define Fc   3
#define Tc   12
#define Dc   64
#define AHc  4
#define DKc  32
#define DVc  32
#define GHc  8
#define Ec   524288
#define BNc  (Bc*Nc)       // 32768
#define Rc   (Bc*Fc*Tc)    // 288 rows of [*,4096]
#define QKN  (AHc*DKc)     // 128
#define GTN  (GHc*Tc)      // 96
#define ETOT (Ec+BNc)      // 557056 edges incl self-loops

typedef __attribute__((ext_vector_type(16))) _Float16 v16h;
typedef __attribute__((ext_vector_type(8)))  _Float16 v8h;
typedef __attribute__((ext_vector_type(8)))  float    v8f;

// ---------------- weight convert/transpose: src[K,N] f32 -> dst[N,K] f16 ----
__global__ void gboot_transpose16(const float* __restrict__ src,
                                  _Float16* __restrict__ dst, int K, int N) {
    int gid = blockIdx.x * blockDim.x + threadIdx.x;
    if (gid >= K * N) return;
    int k = gid / N, n = gid - k * N;
    dst[(size_t)n * K + k] = (_Float16)src[gid];
}

// ---------------- WMMA GEMM: C[M,N] = A[M,K] * B[K,N] (+bias[n]) ------------
// A row-major f16, Bt = B transposed [N,K] f16. One wave per 16x16 C tile.
__global__ void gboot_wmma_gemm(const _Float16* __restrict__ A,
                                const _Float16* __restrict__ Bt,
                                const float* __restrict__ bias,
                                float* __restrict__ C,
                                int M, int N, int K, int nTiles) {
    int wid = (int)((blockIdx.x * blockDim.x + threadIdx.x) >> 5);
    if (wid >= nTiles) return;                 // wave-uniform: EXEC stays full
    int lane = threadIdx.x & 31;
    int ntn  = N >> 4;
    int tm   = wid / ntn;
    int tn   = wid - tm * ntn;
    int hk   = lane >> 4;                      // lane group 0/1
    int l16  = lane & 15;
    const _Float16* ar = A  + (size_t)(tm * 16 + l16) * K;   // row M = l16
    const _Float16* br = Bt + (size_t)(tn * 16 + l16) * K;   // col N = l16
    v8f acc = {};
    for (int k0 = 0; k0 < K; k0 += 32) {
        union { v16h v; v8h h[2]; } a;
        // A 16x32 f16 layout: elems 0..7 -> K=k0+hk*8+e ; 8..15 -> K=k0+16+hk*8+e
        a.h[0] = *(const v8h*)(ar + k0 + hk * 8);
        a.h[1] = *(const v8h*)(ar + k0 + 16 + hk * 8);
        // B 32x16 f16 layout: elems 0..7 -> K=k0+hk*8+e ; 8..15 -> K=k0+16+hk*8+e
        union { v16h v; v8h h[2]; } bu;
        bu.h[0] = *(const v8h*)(br + k0 + hk * 8);
        bu.h[1] = *(const v8h*)(br + k0 + 16 + hk * 8);
        v16h b = bu.v;
        acc = __builtin_amdgcn_wmma_f32_16x16x32_f16(
                  false, a.v, false, b, (short)0, acc, false, false);
    }
    int ncol = tn * 16 + l16;
    float bv = bias ? bias[ncol] : 0.f;
#pragma unroll
    for (int r = 0; r < 8; ++r) {
        int mrow = tm * 16 + r + 8 * hk;       // C/D layout
        C[(size_t)mrow * N + ncol] = acc[r] + bv;
    }
}

// ---------------- EmbedT + LN over N ----------------------------------------
__global__ void gboot_embedT_ln(const float* __restrict__ data,
                                const float* __restrict__ embT_w,
                                const float* __restrict__ g,
                                const float* __restrict__ b,
                                float* __restrict__ x32,
                                _Float16* __restrict__ x16) {
    int r   = blockIdx.x;                        // (b*F+f)*T+t
    int bi  = r / (Fc * Tc);
    int f   = (r / Tc) % Fc;
    int t   = r % Tc;
    int tid = threadIdx.x;
    __shared__ float s1[256], s2[256];
    float* row = x32 + (size_t)r * Nc;
    float ls = 0.f, lq = 0.f;
    for (int n = tid; n < Nc; n += 256) {
        float v = data[(((size_t)bi * Nc + n) * Fc + f) * Tc + t] + embT_w[t * Nc + n];
        row[n] = v; ls += v; lq += v * v;
    }
    s1[tid] = ls; s2[tid] = lq; __syncthreads();
    for (int st = 128; st > 0; st >>= 1) {
        if (tid < st) { s1[tid] += s1[tid + st]; s2[tid] += s2[tid + st]; }
        __syncthreads();
    }
    float mean = s1[0] / Nc;
    float var  = s2[0] / Nc - mean * mean;
    float rs   = rsqrtf(var + 1e-5f);
    for (int n = tid; n < Nc; n += 256) {
        float v = (row[n] - mean) * rs * g[n] + b[n];
        row[n] = v;
        x16[(size_t)r * Nc + n] = (_Float16)v;
    }
}

// ---------------- tiny temporal attention (T=12), per (b,f,h) ---------------
__global__ void gboot_attention(const float* __restrict__ q,
                                const float* __restrict__ k,
                                const float* __restrict__ v,
                                _Float16* __restrict__ ctx16) {
    int blk = blockIdx.x;                 // B*F*AH = 96
    int h   = blk % AHc;
    int bf  = blk / AHc;
    int tid = threadIdx.x;                // 384
    __shared__ float S[Tc][Tc];
    if (tid < Tc * Tc) {
        int t = tid / Tc, tk = tid % Tc;
        const float* qp = q + (size_t)(bf * Tc + t)  * QKN + h * DKc;
        const float* kp = k + (size_t)(bf * Tc + tk) * QKN + h * DKc;
        float s = 0.f;
        for (int d = 0; d < DKc; ++d) s += qp[d] * kp[d];
        S[t][tk] = s * 0.1767766952966369f;   // 1/sqrt(32)
    }
    __syncthreads();
    if (tid < Tc) {
        float mx = S[tid][0];
        for (int j = 1; j < Tc; ++j) mx = fmaxf(mx, S[tid][j]);
        float sum = 0.f;
        for (int j = 0; j < Tc; ++j) { float e = expf(S[tid][j] - mx); S[tid][j] = e; sum += e; }
        float inv = 1.f / sum;
        for (int j = 0; j < Tc; ++j) S[tid][j] *= inv;
    }
    __syncthreads();
    if (tid < Tc * DVc) {
        int t = tid / DVc, d = tid % DVc;
        float s = 0.f;
        for (int tk = 0; tk < Tc; ++tk)
            s += S[t][tk] * v[(size_t)(bf * Tc + tk) * QKN + h * DVc + d];
        ctx16[(size_t)(bf * Tc + t) * QKN + h * DVc + d] = (_Float16)s;
    }
}

// ---------------- proj + residual + LN(tat) ---------------------------------
__global__ void gboot_resid_ln(const float* __restrict__ proj,
                               const float* __restrict__ resid,
                               const float* __restrict__ g,
                               const float* __restrict__ b,
                               float* __restrict__ x2) {
    int r = blockIdx.x, tid = threadIdx.x;
    __shared__ float s1[256], s2[256];
    float* row = x2 + (size_t)r * Nc;
    const float* pr = proj  + (size_t)r * Nc;
    const float* rr = resid + (size_t)r * Nc;
    float ls = 0.f, lq = 0.f;
    for (int n = tid; n < Nc; n += 256) {
        float v = pr[n] + rr[n];
        row[n] = v; ls += v; lq += v * v;
    }
    s1[tid] = ls; s2[tid] = lq; __syncthreads();
    for (int st = 128; st > 0; st >>= 1) {
        if (tid < st) { s1[tid] += s1[tid + st]; s2[tid] += s2[tid + st]; }
        __syncthreads();
    }
    float mean = s1[0] / Nc;
    float rs   = rsqrtf(s2[0] / Nc - mean * mean + 1e-5f);
    for (int n = tid; n < Nc; n += 256)
        row[n] = (row[n] - mean) * rs * g[n] + b[n];
}

// ---------------- feature_temporal_mixer + EmbedS LN ------------------------
__global__ void gboot_mixer_embS(const float* __restrict__ x2,
                                 const float* __restrict__ mix_w,
                                 const float* __restrict__ mix_b,
                                 const float* __restrict__ embS_w,
                                 const float* __restrict__ g,
                                 const float* __restrict__ bb,
                                 _Float16* __restrict__ xf16) {
    int bn = blockIdx.x;                  // 32768
    int bi = bn / Nc, n = bn % Nc;
    int d  = threadIdx.x;                 // 64
    __shared__ float xs[Fc * Tc];
    __shared__ float red[64];
    if (d < Fc * Tc) {
        int f = d / Tc, t = d % Tc;
        xs[d] = x2[(size_t)((bi * Fc + f) * Tc + t) * Nc + n];
    }
    __syncthreads();
    float acc = mix_b[d];
    for (int f = 0; f < Fc; ++f)
        for (int t = 0; t < Tc; ++t)
            acc += xs[f * Tc + t] * mix_w[(d * Tc + t) * Fc + f];
    acc += embS_w[(size_t)n * Dc + d];
    red[d] = acc; __syncthreads();
    for (int st = 32; st > 0; st >>= 1) { if (d < st) red[d] += red[d + st]; __syncthreads(); }
    float mean = red[0] / Dc; __syncthreads();
    float dv = acc - mean;
    red[d] = dv * dv; __syncthreads();
    for (int st = 32; st > 0; st >>= 1) { if (d < st) red[d] += red[d + st]; __syncthreads(); }
    float v = dv * rsqrtf(red[0] / Dc + 1e-5f) * g[d] + bb[d];
    xf16[(size_t)bn * Dc + d] = (_Float16)v;
}

// ---------------- GAT scatter passes ----------------------------------------
__device__ __forceinline__ unsigned gboot_fmap(float f) {
    unsigned x = __float_as_uint(f);
    return (x & 0x80000000u) ? ~x : (x | 0x80000000u);
}
__device__ __forceinline__ float gboot_funmap(unsigned u) {
    return __uint_as_float((u & 0x80000000u) ? (u & 0x7FFFFFFFu) : ~u);
}

__global__ void gboot_gat_init(unsigned* __restrict__ mmax,
                               float* __restrict__ ssum,
                               float* __restrict__ outacc) {
    int i = blockIdx.x * blockDim.x + threadIdx.x;
    if (i < BNc * GHc) { mmax[i] = gboot_fmap(-__builtin_inff()); ssum[i] = 0.f; }
    if (i < BNc * GTN) outacc[i] = 0.f;
}

__device__ __forceinline__ void gboot_edge(int e, int* src, int* dst, const int* edges) {
    if (e < Ec) { *src = edges[e]; *dst = edges[Ec + e]; }
    else        { *src = *dst = e - Ec; }
}

__global__ void gboot_gat_e(const int* __restrict__ edges,
                            const float* __restrict__ xl,
                            const float* __restrict__ xr,
                            const float* __restrict__ att,
                            float* __restrict__ ebuf,
                            unsigned* __restrict__ mmax) {
    int gid = blockIdx.x * blockDim.x + threadIdx.x;
    if (gid >= ETOT * GHc) return;
    int h = gid & 7, e = gid >> 3, src, dst;
    gboot_edge(e, &src, &dst, edges);
    const float* pl = xl + (size_t)src * GTN + h * Tc;
    const float* pr = xr + (size_t)dst * GTN + h * Tc;
    float s = 0.f;
#pragma unroll
    for (int t = 0; t < Tc; ++t) {
        float v = pl[t] + pr[t];
        v = v > 0.f ? v : 0.2f * v;           // leaky_relu(0.2)
        s += v * att[h * Tc + t];
    }
    ebuf[gid] = s;
    atomicMax(&mmax[dst * GHc + h], gboot_fmap(s));
}

__global__ void gboot_gat_exp(const int* __restrict__ edges,
                              float* __restrict__ ebuf,
                              const unsigned* __restrict__ mmax,
                              float* __restrict__ ssum) {
    int gid = blockIdx.x * blockDim.x + threadIdx.x;
    if (gid >= ETOT * GHc) return;
    int h = gid & 7, e = gid >> 3, src, dst;
    gboot_edge(e, &src, &dst, edges);
    float ex = expf(ebuf[gid] - gboot_funmap(mmax[dst * GHc + h]));
    ebuf[gid] = ex;
    atomicAdd(&ssum[dst * GHc + h], ex);
}

__global__ void gboot_gat_agg(const int* __restrict__ edges,
                              const float* __restrict__ ebuf,
                              const float* __restrict__ ssum,
                              const float* __restrict__ xl,
                              float* __restrict__ outacc) {
    int gid = blockIdx.x * blockDim.x + threadIdx.x;
    if (gid >= ETOT * GHc) return;
    int h = gid & 7, e = gid >> 3, src, dst;
    gboot_edge(e, &src, &dst, edges);
    float alpha = ebuf[gid] / ssum[dst * GHc + h];
    const float* pl = xl     + (size_t)src * GTN + h * Tc;
    float*       po = outacc + (size_t)dst * GTN + h * Tc;
#pragma unroll
    for (int t = 0; t < Tc; ++t)
        atomicAdd(&po[t], pl[t] * alpha);
}

// ---------------- fused tail: bias + GLU convs + fc + residual + LN(GH) -----
__global__ void gboot_tail(const float* __restrict__ outacc,
                           const float* __restrict__ gat_bias,
                           const float* __restrict__ g3w, const float* __restrict__ g3b,
                           const float* __restrict__ g5w, const float* __restrict__ g5b,
                           const float* __restrict__ g7w, const float* __restrict__ g7b,
                           const float* __restrict__ fcw, const float* __restrict__ fcb,
                           const float* __restrict__ data,
                           const float* __restrict__ resw, const float* __restrict__ resb,
                           const float* __restrict__ lng, const float* __restrict__ lnb,
                           float* __restrict__ out) {
    int bn = blockIdx.x;                     // 32768
    int bi = bn / Nc, n = bn % Nc;
    int tid = threadIdx.x;                   // 192
    __shared__ float rg[GHc][Tc];
    __shared__ float sg[GHc][24];
    __shared__ float sxv[Tc][GHc];
    __shared__ float mt[Tc], vt[Tc];
    if (tid < GHc * Tc) {
        int h = tid / Tc, t = tid % Tc;
        rg[h][t] = outacc[(size_t)bn * GTN + h * Tc + t] + gat_bias[h * Tc + t];
    }
    __syncthreads();
    {   // GLU convs, kernels 3/5/7, output channel h (gate) and h+8 (value)
        int h = tid / 24, j = tid % 24;
        const float* w; const float* wb; int ks, l;
        if (j < 10)      { ks = 3; l = j;      w = g3w; wb = g3b; }
        else if (j < 18) { ks = 5; l = j - 10; w = g5w; wb = g5b; }
        else             { ks = 7; l = j - 18; w = g7w; wb = g7b; }
        float y1 = wb[h], y2 = wb[h + GHc];
        for (int ci = 0; ci < GHc; ++ci)
            for (int jj = 0; jj < ks; ++jj) {
                float xv = rg[ci][l + jj];
                y1 += xv * w[(h * GHc + ci) * ks + jj];
                y2 += xv * w[((h + GHc) * GHc + ci) * ks + jj];
            }
        float sig = 1.f / (1.f + expf(-y1));
        sg[h][j] = y1 * sig * y2;            // silu(y1) * y2
    }
    __syncthreads();
    if (tid < GHc * Tc) {                    // fc + relu + residual 1x1 + relu
        int h = tid / Tc, t = tid % Tc;
        float acc = fcb[t];
        for (int j = 0; j < 24; ++j) acc += sg[h][j] * fcw[j * Tc + t];
        acc = fmaxf(acc, 0.f);
        float xres = resb[h];
        for (int f = 0; f < Fc; ++f)
            xres += data[(((size_t)bi * Nc + n) * Fc + f) * Tc + t] * resw[h * Fc + f];
        sxv[t][h] = fmaxf(xres + acc, 0.f);
    }
    __syncthreads();
    if (tid < Tc) {                          // LN stats over GH
        float m = 0.f;
        for (int h = 0; h < GHc; ++h) m += sxv[tid][h];
        m /= GHc;
        float v = 0.f;
        for (int h = 0; h < GHc; ++h) { float d = sxv[tid][h] - m; v += d * d; }
        mt[tid] = m; vt[tid] = v / GHc;
    }
    __syncthreads();
    if (tid < GHc * Tc) {
        int h = tid / Tc, t = tid % Tc;
        float v = (sxv[t][h] - mt[t]) * rsqrtf(vt[t] + 1e-5f) * lng[h] + lnb[h];
        out[(((size_t)bi * Nc + n) * GHc + h) * Tc + t] = v;
    }
}

// ---------------- host side -------------------------------------------------
extern "C" void kernel_launch(void* const* d_in, const int* in_sizes, int n_in,
                              void* d_out, int out_size, void* d_ws, size_t ws_size,
                              hipStream_t stream) {
    const float* data   = (const float*)d_in[0];
    const int*   edges  = (const int*)  d_in[1];
    const float* embT_w = (const float*)d_in[2];
    const float* embT_g = (const float*)d_in[3];
    const float* embT_b = (const float*)d_in[4];
    const float* wq     = (const float*)d_in[5];
    const float* wk     = (const float*)d_in[6];
    const float* wv     = (const float*)d_in[7];
    const float* wo     = (const float*)d_in[8];
    const float* tat_g  = (const float*)d_in[9];
    const float* tat_b  = (const float*)d_in[10];
    const float* mix_w  = (const float*)d_in[11];
    const float* mix_b  = (const float*)d_in[12];
    const float* embS_w = (const float*)d_in[13];
    const float* embS_g = (const float*)d_in[14];
    const float* embS_b = (const float*)d_in[15];
    const float* gat_wl = (const float*)d_in[16];
    const float* gat_bl = (const float*)d_in[17];
    const float* gat_wr = (const float*)d_in[18];
    const float* gat_br = (const float*)d_in[19];
    const float* gat_att= (const float*)d_in[20];
    const float* gat_bias=(const float*)d_in[21];
    const float* g3w = (const float*)d_in[22]; const float* g3b = (const float*)d_in[23];
    const float* g5w = (const float*)d_in[24]; const float* g5b = (const float*)d_in[25];
    const float* g7w = (const float*)d_in[26]; const float* g7b = (const float*)d_in[27];
    const float* fcw = (const float*)d_in[28]; const float* fcb = (const float*)d_in[29];
    const float* resw= (const float*)d_in[30]; const float* resb= (const float*)d_in[31];
    const float* lng = (const float*)d_in[32]; const float* lnb = (const float*)d_in[33];
    float* out = (float*)d_out;

    char* ws = (char*)d_ws;
    size_t off = 0;
    auto take = [&](size_t bytes) -> char* {
        char* p = ws + off; off += bytes; off = (off + 255) & ~(size_t)255; return p;
    };
    float*     x32   = (float*)    take((size_t)Rc * Nc * 4);
    _Float16*  x16   = (_Float16*) take((size_t)Rc * Nc * 2);
    _Float16*  wqT   = (_Float16*) take((size_t)QKN * Nc * 2);
    _Float16*  wkT   = (_Float16*) take((size_t)QKN * Nc * 2);
    _Float16*  wvT   = (_Float16*) take((size_t)QKN * Nc * 2);
    _Float16*  woT   = (_Float16*) take((size_t)Nc * QKN * 2);
    _Float16*  wlT   = (_Float16*) take((size_t)GTN * Dc * 2);
    _Float16*  wrT   = (_Float16*) take((size_t)GTN * Dc * 2);
    float*     qb    = (float*)    take((size_t)Rc * QKN * 4);
    float*     kb    = (float*)    take((size_t)Rc * QKN * 4);
    float*     vb    = (float*)    take((size_t)Rc * QKN * 4);
    _Float16*  ctx16 = (_Float16*) take((size_t)Rc * QKN * 2);
    float*     proj  = (float*)    take((size_t)Rc * Nc * 4);
    float*     x2    = (float*)    take((size_t)Rc * Nc * 4);
    _Float16*  xf16  = (_Float16*) take((size_t)BNc * Dc * 2);
    float*     xl    = (float*)    take((size_t)BNc * GTN * 4);
    float*     xr    = (float*)    take((size_t)BNc * GTN * 4);
    unsigned*  mmax  = (unsigned*) take((size_t)BNc * GHc * 4);
    float*     ssum  = (float*)    take((size_t)BNc * GHc * 4);
    float*     outac = (float*)    take((size_t)BNc * GTN * 4);
    float*     ebuf  = (float*)    take((size_t)ETOT * GHc * 4);
    (void)ws_size; (void)n_in; (void)in_sizes; (void)out_size;

    // 0) weights -> f16 transposed
    gboot_transpose16<<<(Nc*QKN + 255)/256, 256, 0, stream>>>(wq, wqT, Nc, QKN);
    gboot_transpose16<<<(Nc*QKN + 255)/256, 256, 0, stream>>>(wk, wkT, Nc, QKN);
    gboot_transpose16<<<(Nc*QKN + 255)/256, 256, 0, stream>>>(wv, wvT, Nc, QKN);
    gboot_transpose16<<<(QKN*Nc + 255)/256, 256, 0, stream>>>(wo, woT, QKN, Nc);
    gboot_transpose16<<<(Dc*GTN + 255)/256, 256, 0, stream>>>(gat_wl, wlT, Dc, GTN);
    gboot_transpose16<<<(Dc*GTN + 255)/256, 256, 0, stream>>>(gat_wr, wrT, Dc, GTN);

    // 1) EmbedT + LN
    gboot_embedT_ln<<<Rc, 256, 0, stream>>>(data, embT_w, embT_g, embT_b, x32, x16);

    // 2) Q/K/V projections (WMMA): [288,4096] x [4096,128]
    {
        int tiles = (Rc/16) * (QKN/16);      // 144
        int blks  = (tiles + 3) / 4;
        gboot_wmma_gemm<<<blks, 128, 0, stream>>>(x16, wqT, nullptr, qb, Rc, QKN, Nc, tiles);
        gboot_wmma_gemm<<<blks, 128, 0, stream>>>(x16, wkT, nullptr, kb, Rc, QKN, Nc, tiles);
        gboot_wmma_gemm<<<blks, 128, 0, stream>>>(x16, wvT, nullptr, vb, Rc, QKN, Nc, tiles);
    }

    // 3) attention (T=12)
    gboot_attention<<<Bc*Fc*AHc, 384, 0, stream>>>(qb, kb, vb, ctx16);

    // 4) output projection (WMMA): [288,128] x [128,4096]
    {
        int tiles = (Rc/16) * (Nc/16);       // 4608
        gboot_wmma_gemm<<<(tiles+3)/4, 128, 0, stream>>>(ctx16, woT, nullptr, proj, Rc, Nc, QKN, tiles);
    }

    // 5) residual + LN(tat)
    gboot_resid_ln<<<Rc, 256, 0, stream>>>(proj, x32, tat_g, tat_b, x2);

    // 6) mixer + EmbedS LN -> xf16 [32768,64]
    gboot_mixer_embS<<<BNc, 64, 0, stream>>>(x2, mix_w, mix_b, embS_w, embS_g, embS_b, xf16);

    // 7) GAT projections (WMMA): [32768,64] x [64,96] + bias
    {
        int tiles = (BNc/16) * (GTN/16);     // 12288
        gboot_wmma_gemm<<<(tiles+3)/4, 128, 0, stream>>>(xf16, wlT, gat_bl, xl, BNc, GTN, Dc, tiles);
        gboot_wmma_gemm<<<(tiles+3)/4, 128, 0, stream>>>(xf16, wrT, gat_br, xr, BNc, GTN, Dc, tiles);
    }

    // 8) GAT edge softmax + aggregate
    gboot_gat_init<<<(BNc*GTN + 255)/256, 256, 0, stream>>>(mmax, ssum, outac);
    {
        int nth = ETOT * GHc;
        int blks = (nth + 255) / 256;
        gboot_gat_e  <<<blks, 256, 0, stream>>>(edges, xl, xr, gat_att, ebuf, mmax);
        gboot_gat_exp<<<blks, 256, 0, stream>>>(edges, ebuf, mmax, ssum);
        gboot_gat_agg<<<blks, 256, 0, stream>>>(edges, ebuf, ssum, xl, outac);
    }

    // 9) fused tail -> out [B,N,GH,T]
    gboot_tail<<<BNc, 192, 0, stream>>>(outac, gat_bias, g3w, g3b, g5w, g5b, g7w, g7b,
                                        fcw, fcb, data, resw, resb, lng, lnb, out);
}
